// TokenRoutedMLPParallel_63582695850551
// MI455X (gfx1250) — compile-verified, matrix operand order checked by reading, not run on hardware
//
#include <hip/hip_runtime.h>
#include <stdint.h>

typedef __attribute__((ext_vector_type(16))) __bf16 v16bf;
typedef __attribute__((ext_vector_type(8)))  float  v8f;
typedef __attribute__((ext_vector_type(4)))  float  f32x4;
typedef __attribute__((ext_vector_type(4)))  unsigned short u16x4;
typedef unsigned short u16;
typedef unsigned int   u32;
typedef unsigned long long u64;

#define T_TOK 4096
#define T_PAD 4224          // +128 rows so partial-tile async loads never go OOB
#define DHID  2048
#define HEXP  1024
#define NEXP  8

#define BM 128
#define BN 128
#define BK 32

// ---------------- workspace layout (bytes) ----------------
#define WS_CNT   0UL
#define WS_OFFS  64UL
#define WS_TOKE  128UL
#define WS_TOKP  (WS_TOKE + 4UL*T_TOK)
#define WS_PERM  (WS_TOKP + 4UL*T_TOK)
#define WS_XBF   65536UL                                  // [T_PAD][DHID] bf16 (expert-sorted)
#define WS_HBF   (WS_XBF + 2UL*T_PAD*DHID)                // [T_PAD][HEXP] bf16
#define WS_WG    (WS_HBF + 2UL*T_PAD*HEXP)                // [E][n=HEXP][k=DHID] bf16
#define WS_WU    (WS_WG + 2UL*NEXP*DHID*HEXP)
#define WS_WD    (WS_WU + 2UL*NEXP*DHID*HEXP)             // [E][n=DHID][k=HEXP] bf16

// ---------------- helpers ----------------
__device__ __forceinline__ u16 bf16rn(float f) {
  u32 x = __float_as_uint(f);
  x += 0x7FFFu + ((x >> 16) & 1u);      // round-to-nearest-even
  return (u16)(x >> 16);
}

// CDNA5 async DMA: global -> LDS, 16 bytes per lane, tracked by ASYNCcnt.
__device__ __forceinline__ void async_cp16(u32 lds_addr, const void* gsrc) {
  asm volatile("global_load_async_to_lds_b128 %0, %1, off"
               :: "v"(lds_addr), "v"((u64)(uintptr_t)gsrc) : "memory");
}
#define WAIT_ASYNC(n) asm volatile("s_wait_asynccnt " #n ::: "memory")

__device__ __forceinline__ v8f v8f_zero() {
  v8f z;
#pragma unroll
  for (int k = 0; k < 8; ++k) z[k] = 0.f;
  return z;
}

__device__ __forceinline__ v8f wmma_bf16(v16bf a, v16bf b, v8f c) {
  return __builtin_amdgcn_wmma_f32_16x16x32_bf16(false, a, false, b, (short)0, c,
                                                 false, false);
}

// Load a 16x32 bf16 fragment from an LDS tile stored row-major [rows][BK].
// A layout (16-bit, 16x32): lanes 0-15 row=l, VGPR0-3 K=0..7, VGPR4-7 K=16..23;
// lanes 16-31 row=l, K=8..15 / 24..31.  B (32x16) mirrors with lane->N since the
// tile is stored N-major [n][k].
__device__ __forceinline__ v16bf ld_frag(const u16* tile, int r16, int lane) {
  const int l = lane & 15, hi = lane >> 4;
  const u16* p = tile + (size_t)(r16 + l) * BK + hi * 8;
  union { f32x4 q[2]; v16bf v; } u;
  u.q[0] = *(const f32x4*)(p);          // K = hi*8 .. hi*8+7        (ds_load_b128)
  u.q[1] = *(const f32x4*)(p + 16);     // K = 16+hi*8 .. 16+hi*8+7  (ds_load_b128)
  return u.v;
}

// ---------------- routing ----------------
__global__ void k_zero(int* cnt) { if (threadIdx.x < NEXP) cnt[threadIdx.x] = 0; }

__global__ void k_route(const long long* __restrict__ ids, int* cnt,
                        int* tok_e, int* tok_p) {
  int t = blockIdx.x * 256 + threadIdx.x;
  long long id = ids[t];
  if (id < 0) id = 0;
  if (id > 99999) id = 99999;
  int e = (int)(id / 12500);
  if (e > 7) e = 7;
  tok_e[t] = e;
  tok_p[t] = atomicAdd(&cnt[e], 1);
}

__global__ void k_scan(const int* cnt, int* offs) {
  if (threadIdx.x == 0) {
    int s = 0;
    for (int i = 0; i < NEXP; ++i) { offs[i] = s; s += cnt[i]; }
    offs[NEXP] = s;
  }
}

__global__ void k_perm(const int* tok_e, const int* tok_p, const int* offs,
                       int* perm) {
  int t = blockIdx.x * 256 + threadIdx.x;
  perm[offs[tok_e[t]] + tok_p[t]] = t;
}

// ---------------- gather + fp32->bf16 ----------------
__global__ __launch_bounds__(256) void k_gather(const float* __restrict__ hs,
                                                const int* __restrict__ perm,
                                                u16* __restrict__ Xbf) {
  const int row = blockIdx.x;
  const int tok = perm[row];
  const int c = blockIdx.y * 1024 + threadIdx.x * 4;
  f32x4 v = *(const f32x4*)(hs + (size_t)tok * DHID + c);
  u16x4 o;
  o.x = bf16rn(v.x); o.y = bf16rn(v.y); o.z = bf16rn(v.z); o.w = bf16rn(v.w);
  *(u16x4*)(Xbf + (size_t)row * DHID + c) = o;
}

// fp32 [R][C] -> bf16 [C][R] (N-major weights), one matrix per blockIdx.z
__global__ __launch_bounds__(256) void k_tcvt(const float* __restrict__ src,
                                              u16* __restrict__ dst,
                                              int R, int C) {
  __shared__ float tile[32][33];
  const float* s = src + (size_t)blockIdx.z * R * C;
  u16* d = dst + (size_t)blockIdx.z * R * C;
  const int c0 = blockIdx.x * 32, r0 = blockIdx.y * 32;
  const int tx = threadIdx.x & 31, ty = threadIdx.x >> 5;
#pragma unroll
  for (int i = 0; i < 4; ++i)
    tile[ty + i * 8][tx] = s[(size_t)(r0 + ty + i * 8) * C + c0 + tx];
  __syncthreads();
#pragma unroll
  for (int i = 0; i < 4; ++i)
    d[(size_t)(c0 + ty + i * 8) * R + r0 + tx] = bf16rn(tile[tx][ty + i * 8]);
}

// ---------------- GEMM1: h = silu(X@Wg) * (X@Wu) ----------------
__global__ __launch_bounds__(256) void k_gemm1(const u16* __restrict__ Xbf,
                                               const u16* __restrict__ WG,
                                               const u16* __restrict__ WU,
                                               const int* __restrict__ offs,
                                               u16* __restrict__ Hbf) {
  __shared__ u16 As[2][BM * BK];
  __shared__ u16 Bg[2][BN * BK];
  __shared__ u16 Bu[2][BN * BK];
  const int e = blockIdx.z;
  const int rowEnd = offs[e + 1];
  const int row0 = offs[e] + blockIdx.y * BM;
  if (row0 >= rowEnd) return;
  const int n0 = blockIdx.x * BN;
  const u16* wg = WG + (size_t)e * DHID * HEXP + (size_t)n0 * DHID;
  const u16* wu = WU + (size_t)e * DHID * HEXP + (size_t)n0 * DHID;
  const u16* xa = Xbf + (size_t)row0 * DHID;
  const int tid = threadIdx.x;
  const int lane = tid & 31, wid = tid >> 5;
  const int wm = (wid & 3) * 32, wn = (wid >> 2) * 64;

  v8f accg[2][4], accu[2][4];
#pragma unroll
  for (int i = 0; i < 2; ++i)
#pragma unroll
    for (int j = 0; j < 4; ++j) { accg[i][j] = v8f_zero(); accu[i][j] = v8f_zero(); }

  auto issue = [&](int buf, int k0) {     // 6 async b128 per thread per tile
    const int q = tid & 3, m = tid >> 2;
#pragma unroll
    for (int r = 0; r < 2; ++r) {
      const int row = m + 64 * r;
      async_cp16((u32)(uintptr_t)&As[buf][row * BK + q * 8],
                 xa + (size_t)row * DHID + k0 + q * 8);
      async_cp16((u32)(uintptr_t)&Bg[buf][row * BK + q * 8],
                 wg + (size_t)row * DHID + k0 + q * 8);
      async_cp16((u32)(uintptr_t)&Bu[buf][row * BK + q * 8],
                 wu + (size_t)row * DHID + k0 + q * 8);
    }
  };

  issue(0, 0);
  const int KT = DHID / BK;
  for (int kt = 0; kt < KT; ++kt) {
    const int cur = kt & 1;
    if (kt + 1 < KT) { issue(cur ^ 1, (kt + 1) * BK); WAIT_ASYNC(6); }
    else             { WAIT_ASYNC(0); }
    __syncthreads();
    v16bf af[2], bgf[4], buf_[4];
#pragma unroll
    for (int i = 0; i < 2; ++i) af[i] = ld_frag(&As[cur][0], wm + i * 16, lane);
#pragma unroll
    for (int j = 0; j < 4; ++j) bgf[j] = ld_frag(&Bg[cur][0], wn + j * 16, lane);
#pragma unroll
    for (int j = 0; j < 4; ++j) buf_[j] = ld_frag(&Bu[cur][0], wn + j * 16, lane);
#pragma unroll
    for (int i = 0; i < 2; ++i)
#pragma unroll
      for (int j = 0; j < 4; ++j) {
        accg[i][j] = wmma_bf16(af[i], bgf[j], accg[i][j]);
        accu[i][j] = wmma_bf16(af[i], buf_[j], accu[i][j]);
      }
    __syncthreads();
  }

  // epilogue: silu(g)*u -> bf16.  C layout: VGPR r, lanes0-15 M=r, lanes16-31 M=r+8.
  const int l = lane & 15, hi = lane >> 4;
#pragma unroll
  for (int i = 0; i < 2; ++i)
#pragma unroll
    for (int j = 0; j < 4; ++j) {
      const int col = n0 + wn + j * 16 + l;
#pragma unroll
      for (int r = 0; r < 8; ++r) {
        const int row = row0 + wm + i * 16 + hi * 8 + r;
        if (row < rowEnd) {
          float g = accg[i][j][r], u = accu[i][j][r];
          float h = (g / (1.f + __expf(-g))) * u;
          Hbf[(size_t)row * HEXP + col] = bf16rn(h);
        }
      }
    }
}

// ---------------- GEMM2: out[perm[row]] = H @ Wd ----------------
__global__ __launch_bounds__(256) void k_gemm2(const u16* __restrict__ Hbf,
                                               const u16* __restrict__ WD,
                                               const int* __restrict__ offs,
                                               const int* __restrict__ perm,
                                               float* __restrict__ out) {
  __shared__ u16 As[2][BM * BK];
  __shared__ u16 Bd[2][BN * BK];
  const int e = blockIdx.z;
  const int rowEnd = offs[e + 1];
  const int row0 = offs[e] + blockIdx.y * BM;
  if (row0 >= rowEnd) return;
  const int n0 = blockIdx.x * BN;
  const u16* wd = WD + (size_t)e * DHID * HEXP + (size_t)n0 * HEXP;
  const u16* ha = Hbf + (size_t)row0 * HEXP;
  const int tid = threadIdx.x;
  const int lane = tid & 31, wid = tid >> 5;
  const int wm = (wid & 3) * 32, wn = (wid >> 2) * 64;

  v8f acc[2][4];
#pragma unroll
  for (int i = 0; i < 2; ++i)
#pragma unroll
    for (int j = 0; j < 4; ++j) acc[i][j] = v8f_zero();

  auto issue = [&](int buf, int k0) {     // 4 async b128 per thread per tile
    const int q = tid & 3, m = tid >> 2;
#pragma unroll
    for (int r = 0; r < 2; ++r) {
      const int row = m + 64 * r;
      async_cp16((u32)(uintptr_t)&As[buf][row * BK + q * 8],
                 ha + (size_t)row * HEXP + k0 + q * 8);
      async_cp16((u32)(uintptr_t)&Bd[buf][row * BK + q * 8],
                 wd + (size_t)row * HEXP + k0 + q * 8);
    }
  };

  issue(0, 0);
  const int KT = HEXP / BK;
  for (int kt = 0; kt < KT; ++kt) {
    const int cur = kt & 1;
    if (kt + 1 < KT) { issue(cur ^ 1, (kt + 1) * BK); WAIT_ASYNC(4); }
    else             { WAIT_ASYNC(0); }
    __syncthreads();
    v16bf af[2], bf[4];
#pragma unroll
    for (int i = 0; i < 2; ++i) af[i] = ld_frag(&As[cur][0], wm + i * 16, lane);
#pragma unroll
    for (int j = 0; j < 4; ++j) bf[j] = ld_frag(&Bd[cur][0], wn + j * 16, lane);
#pragma unroll
    for (int i = 0; i < 2; ++i)
#pragma unroll
      for (int j = 0; j < 4; ++j) acc[i][j] = wmma_bf16(af[i], bf[j], acc[i][j]);
    __syncthreads();
  }

  const int l = lane & 15, hi = lane >> 4;
#pragma unroll
  for (int i = 0; i < 2; ++i)
#pragma unroll
    for (int j = 0; j < 4; ++j) {
      const int col = n0 + wn + j * 16 + l;
#pragma unroll
      for (int r = 0; r < 8; ++r) {
        const int row = row0 + wm + i * 16 + hi * 8 + r;
        if (row < rowEnd) {
          const int tok = perm[row];
          out[(size_t)tok * DHID + col] = acc[i][j][r];
        }
      }
    }
}

// ---------------- host launcher ----------------
extern "C" void kernel_launch(void* const* d_in, const int* in_sizes, int n_in,
                              void* d_out, int out_size, void* d_ws, size_t ws_size,
                              hipStream_t stream) {
  const float* hidden      = (const float*)d_in[0];
  const long long* tokids  = (const long long*)d_in[1];
  const float* gate        = (const float*)d_in[2];
  const float* up          = (const float*)d_in[3];
  const float* down        = (const float*)d_in[4];
  float* out               = (float*)d_out;
  char* ws                 = (char*)d_ws;

  int* cnt   = (int*)(ws + WS_CNT);
  int* offs  = (int*)(ws + WS_OFFS);
  int* tok_e = (int*)(ws + WS_TOKE);
  int* tok_p = (int*)(ws + WS_TOKP);
  int* perm  = (int*)(ws + WS_PERM);
  u16* Xbf   = (u16*)(ws + WS_XBF);
  u16* Hbf   = (u16*)(ws + WS_HBF);
  u16* WG    = (u16*)(ws + WS_WG);
  u16* WU    = (u16*)(ws + WS_WU);
  u16* WD    = (u16*)(ws + WS_WD);

  k_zero<<<1, 32, 0, stream>>>(cnt);
  k_route<<<T_TOK / 256, 256, 0, stream>>>(tokids, cnt, tok_e, tok_p);
  k_scan<<<1, 32, 0, stream>>>(cnt, offs);
  k_perm<<<T_TOK / 256, 256, 0, stream>>>(tok_e, tok_p, offs, perm);
  k_gather<<<dim3(T_TOK, DHID / 1024), 256, 0, stream>>>(hidden, perm, Xbf);
  // gate/up: [K=2048][N=1024] -> [N][K];  down: [K=1024][N=2048] -> [N][K]
  k_tcvt<<<dim3(HEXP / 32, DHID / 32, NEXP), 256, 0, stream>>>(gate, WG, DHID, HEXP);
  k_tcvt<<<dim3(HEXP / 32, DHID / 32, NEXP), 256, 0, stream>>>(up,   WU, DHID, HEXP);
  k_tcvt<<<dim3(DHID / 32, HEXP / 32, NEXP), 256, 0, stream>>>(down, WD, HEXP, DHID);
  k_gemm1<<<dim3(HEXP / BN, T_TOK / BM, NEXP), 256, 0, stream>>>(Xbf, WG, WU, offs, Hbf);
  k_gemm2<<<dim3(DHID / BN, T_TOK / BM, NEXP), 256, 0, stream>>>(Hbf, WD, offs, perm, out);
}
